// SelfAttention_10342281248792
// MI455X (gfx1250) — matrix operand order repaired and checked by reading, not compiled, not run on hardware
//
#include <hip/hip_runtime.h>

// MI455X / gfx1250 fused self-attention:
//   kernel 1: QKV projection (f16 WMMA, f32 acc) + RoPE, V stored transposed
//   kernel 2: flash attention, K/V tiles staged to LDS by the Tensor Data
//             Mover (double buffered, TENSORcnt-synced), f16 WMMA, f32 softmax

typedef _Float16 v8h  __attribute__((ext_vector_type(8)));
typedef _Float16 v16h __attribute__((ext_vector_type(16)));
typedef float    v8f  __attribute__((ext_vector_type(8)));
typedef unsigned int u32x4 __attribute__((ext_vector_type(4)));
typedef int          i32x4 __attribute__((ext_vector_type(4)));
typedef int          i32x8 __attribute__((ext_vector_type(8)));

#define S_LEN 2048
#define HID_D 1024
#define NHEAD 16
#define HDIM  64
#define KV_CH 64          // kv positions per flash iteration
#define LDS_STRIDE 72     // 64 halfs + 8 pad halfs (4 dwords) -> conflict-free

#if defined(__has_builtin)
#if __has_builtin(__builtin_amdgcn_tensor_load_to_lds) && __has_builtin(__builtin_amdgcn_s_wait_tensorcnt)
#define USE_TDM 1
#endif
#endif
#ifndef USE_TDM
#define USE_TDM 0
#endif

static __device__ inline v16h cat8(v8h lo, v8h hi) {
  return __builtin_shufflevector(lo, hi, 0,1,2,3,4,5,6,7,8,9,10,11,12,13,14,15);
}
// B-fragment: lane = N (l&15); 16 contiguous halves along K, +16 K for lanes 16-31
static __device__ inline v16h load_b16(const _Float16* p) {
  v8h lo = *(const v8h*)p;
  v8h hi = *(const v8h*)(p + 8);
  return cat8(lo, hi);
}
// A-fragment: lane l holds M=l&15; halves i -> K=(l>=16?16:0)+i
static __device__ inline v16h load_afrag(const _Float16* row, int hi8) {
  v8h lo = *(const v8h*)(row + hi8 * 16);
  v8h hi = *(const v8h*)(row + hi8 * 16 + 8);
  return cat8(lo, hi);
}
static __device__ inline v8f wmma_f16(v16h a, v16h b, v8f c) {
  return __builtin_amdgcn_wmma_f32_16x16x32_f16(false, a, false, b, (short)0, c, false, false);
}

#if USE_TDM
// Tensor DMA: 2D tile (tile1 rows x tile0 elems, 2-byte elems) global -> LDS,
// LDS pad of 4 dwords after each 32-dword (64-half) row => LDS_STRIDE halfs.
static __device__ inline void tdm_load_2d(unsigned lds_addr, const void* gptr,
                                          unsigned dim0, unsigned dim1,
                                          unsigned tile0, unsigned tile1,
                                          unsigned long long stride0) {
  unsigned long long ga = (unsigned long long)(uintptr_t)gptr;
  u32x4 g0;
  g0[0] = 1u;                                           // count=1 (user D#)
  g0[1] = lds_addr;                                     // LDS byte address
  g0[2] = (unsigned)(ga & 0xFFFFFFFFu);                 // global addr [31:0]
  g0[3] = (unsigned)((ga >> 32) & 0x01FFFFFFu)          // global addr [56:32]
        | 0x80000000u;                                  // type=2 ("image")
  i32x8 g1;
  g1[0] = (int)((1u << 16)      // data_size = 2 bytes
              | (1u << 20)      // pad_enable
              | (4u << 22)      // pad_interval: 32 dwords
              | (3u << 25));    // pad_amount: 4 dwords
  g1[1] = (int)((dim0 & 0xFFFFu) << 16);                        // tensor_dim0 lo
  g1[2] = (int)(((dim0 >> 16) & 0xFFFFu) | ((dim1 & 0xFFFFu) << 16));
  g1[3] = (int)(((dim1 >> 16) & 0xFFFFu) | ((tile0 & 0xFFFFu) << 16));
  g1[4] = (int)(tile1 & 0xFFFFu);                               // tile_dim1, tile_dim2=0
  g1[5] = (int)(stride0 & 0xFFFFFFFFull);                       // tensor_dim0_stride lo
  g1[6] = (int)((stride0 >> 32) & 0xFFFFull);                   // stride hi, dim1_stride=0
  g1[7] = 0;
  i32x4 z4 = {0, 0, 0, 0};
  i32x8 z8 = {0, 0, 0, 0, 0, 0, 0, 0};
  // clang-23 / amdgpu-toolchain 6-arg form: (g0, g1, g2, g3, extra, cpol)
  __builtin_amdgcn_tensor_load_to_lds(g0, g1, z4, z4, z8, 0);
}
static __device__ inline unsigned lds_byte_addr(const void* p) {
  return (unsigned)(unsigned long long)(uintptr_t)p;  // flat LDS addr[31:0] = offset
}
#endif

// -------------------------------------------------------------------------
// Kernel 1: per block -> 64 tokens x one full head (64 cols) of one of Q/K/V.
// GEMM: out[m][d] = sum_e hidden[m][e] * W[d][e]   (A = hidden, B = W rows)
// -------------------------------------------------------------------------
__global__ __launch_bounds__(256) void qkv_rope_kernel(
    const float* __restrict__ hid, const float* __restrict__ Wq,
    const float* __restrict__ Wk,  const float* __restrict__ Wv,
    _Float16* __restrict__ qws, _Float16* __restrict__ kws,
    _Float16* __restrict__ vtws)
{
  __shared__ _Float16 lA[64 * 40];   // hidden tile 64x32 f16, padded stride 40
  __shared__ _Float16 lB[64 * 40];   // weight tile 64x32 f16
  __shared__ float    xt[64 * 65];   // f32 result tile for RoPE / transpose

  const int t    = threadIdx.x;
  const int l    = t & 31;
  const int w    = t >> 5;
  const int hi8  = (l >> 4) & 1;
  const int ln16 = l & 15;
  const int mt   = w & 3;            // wave's 16-row M tile (0..3)
  const int nt0  = (w >> 2) * 2;     // wave's two 16-col N tiles

  const int m0   = blockIdx.x * 64;  // token tile base (over B*S)
  const int h    = blockIdx.y;       // head
  const int mat  = blockIdx.z;       // 0=Q 1=K 2=V
  const float* W = (mat == 0) ? Wq : (mat == 1) ? Wk : Wv;
  const int d0   = h * HDIM;

  v8f acc0 = {0.f,0.f,0.f,0.f,0.f,0.f,0.f,0.f};
  v8f acc1 = acc0;

  const int lr = t >> 2;             // coop-load row 0..63
  const int lc = (t & 3) * 8;        // coop-load col 0,8,16,24

#pragma unroll 1
  for (int k0 = 0; k0 < HID_D; k0 += 32) {
    __syncthreads();
    {
      const float* gp = hid + (size_t)(m0 + lr) * HID_D + k0 + lc;
      float4 f0 = *(const float4*)gp;
      float4 f1 = *(const float4*)(gp + 4);
      v8h h8;
      h8[0]=(_Float16)f0.x; h8[1]=(_Float16)f0.y; h8[2]=(_Float16)f0.z; h8[3]=(_Float16)f0.w;
      h8[4]=(_Float16)f1.x; h8[5]=(_Float16)f1.y; h8[6]=(_Float16)f1.z; h8[7]=(_Float16)f1.w;
      *(v8h*)&lA[lr * 40 + lc] = h8;
    }
    {
      const float* gp = W + (size_t)(d0 + lr) * HID_D + k0 + lc;
      float4 f0 = *(const float4*)gp;
      float4 f1 = *(const float4*)(gp + 4);
      v8h h8;
      h8[0]=(_Float16)f0.x; h8[1]=(_Float16)f0.y; h8[2]=(_Float16)f0.z; h8[3]=(_Float16)f0.w;
      h8[4]=(_Float16)f1.x; h8[5]=(_Float16)f1.y; h8[6]=(_Float16)f1.z; h8[7]=(_Float16)f1.w;
      *(v8h*)&lB[lr * 40 + lc] = h8;
    }
    __syncthreads();
    v16h a  = load_afrag(&lA[(mt * 16 + ln16) * 40], hi8);
    v16h b0 = load_b16(&lB[(nt0 * 16 + ln16) * 40 + hi8 * 16]);
    v16h b1 = load_b16(&lB[((nt0 + 1) * 16 + ln16) * 40 + hi8 * 16]);
    acc0 = wmma_f16(a, b0, acc0);
    acc1 = wmma_f16(a, b1, acc1);
  }

#pragma unroll
  for (int r = 0; r < 8; ++r) {
    int row = mt * 16 + r + 8 * hi8;
    xt[row * 65 + nt0 * 16 + ln16]       = acc0[r];
    xt[row * 65 + (nt0 + 1) * 16 + ln16] = acc1[r];
  }
  __syncthreads();

  if (mat < 2) {
    _Float16* dst = (mat == 0) ? qws : kws;
    const int r   = t >> 2;
    const int c0  = (t & 3) * 16;
    const int m   = m0 + r;
    const int b   = m >> 11;            // S=2048
    const int pos = m & (S_LEN - 1);
    v16h o;
#pragma unroll
    for (int j = 0; j < 16; ++j) {
      int   c   = c0 + j;
      float x   = xt[r * 65 + c];
      float xp  = xt[r * 65 + ((c < 32) ? c + 32 : c - 32)];
      float ang = (float)pos * __powf(10000.0f, -(float)(c & 31) * (1.0f / 32.0f));
      float sn, cs;
      __sincosf(ang, &sn, &cs);
      o[j] = (_Float16)(x * cs + ((c < 32) ? -xp : xp) * sn);
    }
    *(v16h*)&dst[(((size_t)b * NHEAD + h) * S_LEN + pos) * HDIM + c0] = o;
  } else {
    const int dd  = t >> 2;
    const int p0  = (t & 3) * 16;
    const int m   = m0 + p0;
    const int b   = m >> 11;
    const int pos = m & (S_LEN - 1);
    v16h o;
#pragma unroll
    for (int j = 0; j < 16; ++j)
      o[j] = (_Float16)xt[(p0 + j) * 65 + dd];
    *(v16h*)&vtws[(((size_t)b * NHEAD + h) * HDIM + dd) * S_LEN + pos] = o;
  }
}

// -------------------------------------------------------------------------
// Kernel 2: flash attention. Block = (b, h, 128 q rows); wave = 16 q rows.
// K/V 64x64 tiles staged to LDS by TDM (double buffered), 16 wmma / chunk.
// -------------------------------------------------------------------------
__global__ __launch_bounds__(256) void flash_attn_kernel(
    const _Float16* __restrict__ qws, const _Float16* __restrict__ kws,
    const _Float16* __restrict__ vtws, const float* __restrict__ mask,
    float* __restrict__ out)
{
  __shared__ _Float16 kbuf[2][64 * LDS_STRIDE];   // K tile: row=kv, col=d
  __shared__ _Float16 vbuf[2][64 * LDS_STRIDE];   // V tile: row=d, col=kv
  __shared__ _Float16 lp[8][16 * LDS_STRIDE];     // per-wave prob tile 16x64

  const int t    = threadIdx.x;
  const int l    = t & 31;
  const int w    = t >> 5;
  const int hi8  = (l >> 4) & 1;
  const int ln16 = l & 15;

  const int q0 = blockIdx.x * 128 + w * 16;
  const int h  = blockIdx.y;
  const int b  = blockIdx.z;

  const size_t bh = (size_t)b * NHEAD + h;
  const _Float16* qbase = qws  + bh * S_LEN * HDIM;
  const _Float16* kbase = kws  + bh * S_LEN * HDIM;   // [S][64]
  const _Float16* vbase = vtws + bh * HDIM * S_LEN;   // [64][S]
  const float*    mrow  = mask + (size_t)b * S_LEN;

  const _Float16* qrow = qbase + (size_t)(q0 + ln16) * HDIM;
  v16h aq0 = load_afrag(qrow, hi8);
  v16h aq1 = load_afrag(qrow + 32, hi8);

  v8f zero = {0.f,0.f,0.f,0.f,0.f,0.f,0.f,0.f};
  v8f acc[4];
  float mrun[8], psum[8];     // psum: lane-local partial row sum
#pragma unroll
  for (int r = 0; r < 8; ++r) { mrun[r] = -3.0e38f; psum[r] = 0.f; }
#pragma unroll
  for (int dt = 0; dt < 4; ++dt) acc[dt] = zero;

  _Float16* myp = &lp[w][0];
  const int NIT = S_LEN / KV_CH;

#if USE_TDM
  if (w == 0) {
    tdm_load_2d(lds_byte_addr(&kbuf[0][0]), kbase, HDIM, S_LEN, 64, 64, HDIM);
    tdm_load_2d(lds_byte_addr(&vbuf[0][0]), vbase, S_LEN, HDIM, 64, 64, S_LEN);
  }
#endif

#pragma unroll 1
  for (int it = 0; it < NIT; ++it) {
    const int kv0 = it * KV_CH;
#if USE_TDM
    const int cur = it & 1;
    if (w == 0) {
      if (it + 1 < NIT) {
        const int kvn = kv0 + KV_CH;
        tdm_load_2d(lds_byte_addr(&kbuf[cur ^ 1][0]),
                    kbase + (size_t)kvn * HDIM, HDIM, S_LEN, 64, 64, HDIM);
        tdm_load_2d(lds_byte_addr(&vbuf[cur ^ 1][0]),
                    vbase + kvn, S_LEN, HDIM, 64, 64, S_LEN);
        __builtin_amdgcn_s_wait_tensorcnt(2);   // current tile's 2 DMAs done
      } else {
        __builtin_amdgcn_s_wait_tensorcnt(0);
      }
    }
    __syncthreads();
#else
    const int cur = 0;
    __syncthreads();
    {
      const int rr = t >> 2, cc = (t & 3) * 16;
      *(v16h*)&kbuf[0][rr * LDS_STRIDE + cc] =
          *(const v16h*)(kbase + (size_t)(kv0 + rr) * HDIM + cc);
      *(v16h*)&vbuf[0][rr * LDS_STRIDE + cc] =
          *(const v16h*)(vbase + (size_t)rr * S_LEN + kv0 + cc);
    }
    __syncthreads();
#endif
    const _Float16* kb = &kbuf[cur][0];
    const _Float16* vb = &vbuf[cur][0];

    // ---- scores: Q(16x64) @ K^T(64x64) -> four 16x16 tiles, 8 wmma
    v8f s[4] = {zero, zero, zero, zero};
#pragma unroll
    for (int n = 0; n < 4; ++n) {
      v16h bk0 = load_b16(&kb[(n * 16 + ln16) * LDS_STRIDE + hi8 * 16]);
      v16h bk1 = load_b16(&kb[(n * 16 + ln16) * LDS_STRIDE + 32 + hi8 * 16]);
      s[n] = wmma_f16(aq0, bk0, s[n]);
      s[n] = wmma_f16(aq1, bk1, s[n]);
    }

    // ---- online softmax; only row-max needs a lane tree
    float mk[4];
#pragma unroll
    for (int n = 0; n < 4; ++n) mk[n] = mrow[kv0 + n * 16 + ln16];

#pragma unroll
    for (int r = 0; r < 8; ++r) {
      float x0 = s[0][r] * 0.125f + mk[0];
      float x1 = s[1][r] * 0.125f + mk[1];
      float x2 = s[2][r] * 0.125f + mk[2];
      float x3 = s[3][r] * 0.125f + mk[3];
      float mx = fmaxf(fmaxf(x0, x1), fmaxf(x2, x3));
      mx = fmaxf(mx, __shfl_xor(mx, 1));
      mx = fmaxf(mx, __shfl_xor(mx, 2));
      mx = fmaxf(mx, __shfl_xor(mx, 4));
      mx = fmaxf(mx, __shfl_xor(mx, 8));
      float mnew = fmaxf(mrun[r], mx);
      float f    = __expf(mrun[r] - mnew);
      mrun[r] = mnew;
      acc[0][r] *= f; acc[1][r] *= f; acc[2][r] *= f; acc[3][r] *= f;
      float e0 = __expf(x0 - mnew);
      float e1 = __expf(x1 - mnew);
      float e2 = __expf(x2 - mnew);
      float e3 = __expf(x3 - mnew);
      psum[r] = psum[r] * f + ((e0 + e1) + (e2 + e3));   // lane-local columns
      int row = (r + 8 * hi8) * LDS_STRIDE + ln16;
      myp[row]      = (_Float16)e0;
      myp[row + 16] = (_Float16)e1;
      myp[row + 32] = (_Float16)e2;
      myp[row + 48] = (_Float16)e3;
    }
    asm volatile("s_wait_dscnt 0" ::: "memory");   // wave-local LDS RAW
    v16h ap0 = load_afrag(&myp[ln16 * LDS_STRIDE], hi8);
    v16h ap1 = load_afrag(&myp[ln16 * LDS_STRIDE + 32], hi8);

    // ---- ctx += P(16x64) @ V(64x64): 8 wmma
#pragma unroll
    for (int dt = 0; dt < 4; ++dt) {
      v16h bv0 = load_b16(&vb[(dt * 16 + ln16) * LDS_STRIDE + hi8 * 16]);
      v16h bv1 = load_b16(&vb[(dt * 16 + ln16) * LDS_STRIDE + 32 + hi8 * 16]);
      acc[dt] = wmma_f16(ap0, bv0, acc[dt]);
      acc[dt] = wmma_f16(ap1, bv1, acc[dt]);
    }
    __syncthreads();   // readers done before this buffer is re-filled
  }

  // ---- one deferred 16-lane reduction of the row sums
  float srun[8];
#pragma unroll
  for (int r = 0; r < 8; ++r) {
    float s = psum[r];
    s += __shfl_xor(s, 1);
    s += __shfl_xor(s, 2);
    s += __shfl_xor(s, 4);
    s += __shfl_xor(s, 8);
    srun[r] = s;
  }

#pragma unroll
  for (int dt = 0; dt < 4; ++dt) {
#pragma unroll
    for (int r = 0; r < 8; ++r) {
      int qr = q0 + r + 8 * hi8;
      out[((size_t)b * S_LEN + qr) * HID_D + h * HDIM + dt * 16 + ln16] =
          acc[dt][r] / srun[r];
    }
  }
}

extern "C" void kernel_launch(void* const* d_in, const int* in_sizes, int n_in,
                              void* d_out, int out_size, void* d_ws, size_t ws_size,
                              hipStream_t stream) {
  const float* hid  = (const float*)d_in[0];
  const float* mask = (const float*)d_in[1];
  const float* Wq   = (const float*)d_in[2];
  const float* Wk   = (const float*)d_in[3];
  const float* Wv   = (const float*)d_in[4];
  float* out = (float*)d_out;

  const size_t elems = (size_t)2 * NHEAD * S_LEN * HDIM;  // per-tensor f16 count
  _Float16* qws = (_Float16*)d_ws;
  _Float16* kws = qws + elems;
  _Float16* vt  = kws + elems;

  qkv_rope_kernel<<<dim3((2 * S_LEN) / 64, NHEAD, 3), 256, 0, stream>>>(
      hid, Wq, Wk, Wv, qws, kws, vt);
  flash_attn_kernel<<<dim3(S_LEN / 128, NHEAD, 2), 256, 0, stream>>>(
      qws, kws, vt, mask, out);
}
